// BlockLSTM_71880572666298
// MI455X (gfx1250) — matrix operand order repaired and checked by reading, not compiled, not run on hardware
//
#include <hip/hip_runtime.h>
#include <hip/hip_fp16.h>
#include <hip/hip_bf16.h>

typedef _Float16 half_t;
typedef __attribute__((ext_vector_type(16))) _Float16 v16h;
typedef __attribute__((ext_vector_type(8)))  _Float16 v8h;
typedef __attribute__((ext_vector_type(8)))  float    v8f;
typedef __attribute__((ext_vector_type(4)))  float    v4f;

#define Hd   128      // hidden size (== input size for all layers here)
#define G4   512      // 4*H gates
#define Bx   512      // batch
#define Sx   512      // sequence length
#define TPB  512      // 16 waves of 32
#define ROWS 16       // batch rows per workgroup

// LDS partition (bytes):
//   wih   : [512][128] f16  ->  131072  @ 0
//   whh   : [512][128] f16  ->  131072  @ 131072
//   gates : [16][512]  f32  ->   32768  @ 262144
//   cbuf  : [16][128]  f32  ->    8192  @ 294912
//   hbuf  : [16][128]  f16  ->    4096  @ 303104
#define SMEM_BYTES 307200

__device__ __forceinline__ float fsig(float x) {
    return 1.0f / (1.0f + __expf(-x));
}

// A-fragment (16x32 f16, wave32 layout): lane's row M = lane%16, hi = lane>=16.
// halves e=0..15 : K = k0 + e + 16*hi.
__device__ __forceinline__ v16h load_a_frag(const float* __restrict__ row, int k0, int hi) {
    const float* p1 = row + k0 + hi * 16;
    const float* p2 = row + k0 + 8 + hi * 16;
    v16h a;
#pragma unroll
    for (int e = 0; e < 8; ++e) {
        a[e]     = (half_t)p1[e];
        a[e + 8] = (half_t)p2[e];
    }
    return a;
}
__device__ __forceinline__ v16h load_a_frag(const half_t* __restrict__ row, int k0, int hi) {
    const v8h c1 = *(const v8h*)(row + k0 + hi * 16);
    const v8h c2 = *(const v8h*)(row + k0 + 8 + hi * 16);
    v16h a;
#pragma unroll
    for (int e = 0; e < 8; ++e) {
        a[e]     = c1[e];
        a[e + 8] = c2[e];
    }
    return a;
}

// B-fragment (32x16 f16): lane holds column N = n0 + lane%16,
// K = k0 + 16*hi + e for e=0..15 (contiguous in k -> one 32B LDS read).
__device__ __forceinline__ v16h load_b_frag(const half_t* __restrict__ w, int n, int k0, int hi) {
    return *(const v16h*)(w + (size_t)n * Hd + k0 + hi * 16);
}

template <typename TIn, bool FINAL>
__global__ __launch_bounds__(TPB) void lstm_layer_kernel(
    const TIn* __restrict__ x,        // element (b,t,k) at b*strideB + t*strideT + k
    long long strideB, long long strideT,
    const float* __restrict__ Wih,    // (4H, H) row-major
    const float* __restrict__ Whh,    // (4H, H) row-major
    const float* __restrict__ bih,    // (4H)
    const float* __restrict__ bhh,    // (4H)
    half_t* __restrict__ yh,          // (S,B,H) f16   (used when !FINAL)
    float*  __restrict__ yf)          // (S,H,B) f32   (used when FINAL)
{
    extern __shared__ __align__(16) char smem[];
    half_t* wih   = (half_t*)(smem);
    half_t* whh   = (half_t*)(smem + 131072);
    float*  gates = (float*) (smem + 262144);
    float*  cbuf  = (float*) (smem + 294912);
    half_t* hbuf  = (half_t*)(smem + 303104);

    const int tid  = threadIdx.x;
    const int wave = tid >> 5;
    const int lane = tid & 31;
    const int hi   = lane >> 4;       // 0 or 1
    const int lm   = lane & 15;
    const int b0   = blockIdx.x * ROWS;

    // Stage weights to LDS as f16 (read once, reused 512 timesteps).
    for (int i = tid; i < G4 * Hd; i += TPB) {
        wih[i] = (half_t)Wih[i];
        whh[i] = (half_t)Whh[i];
    }
    for (int i = tid; i < ROWS * Hd; i += TPB) {
        cbuf[i] = 0.0f;
        hbuf[i] = (half_t)0.0f;
    }
    // Each wave owns gate-column tiles [n0, n0+16) and [n0+16, n0+32).
    const int   n0    = wave * 32;
    const float biasA = bih[n0 + lm]      + bhh[n0 + lm];
    const float biasB = bih[n0 + 16 + lm] + bhh[n0 + 16 + lm];
    __syncthreads();

    const int  m    = lm;                                  // A/D row handled by this lane
    const TIn* xrow = x + (long long)(b0 + m) * strideB;   // + t*strideT + k

    for (int t = 0; t < Sx; ++t) {
        v8f accA, accB;
#pragma unroll
        for (int r = 0; r < 8; ++r) { accA[r] = biasA; accB[r] = biasB; }

        // Input contribution: gates += x_t @ Wih^T   (K = 128)
        const TIn* xt = xrow + (long long)t * strideT;
#pragma unroll
        for (int kk = 0; kk < 4; ++kk) {
            const int k0 = kk * 32;
            const v16h a  = load_a_frag(xt, k0, hi);
            const v16h bA = load_b_frag(wih, n0 + lm,      k0, hi);
            const v16h bB = load_b_frag(wih, n0 + 16 + lm, k0, hi);
            accA = __builtin_amdgcn_wmma_f32_16x16x32_f16(false, a, false, bA, (short)0, accA, false, false);
            accB = __builtin_amdgcn_wmma_f32_16x16x32_f16(false, a, false, bB, (short)0, accB, false, false);
        }
        // Recurrent contribution: gates += h_{t-1} @ Whh^T   (K = 128)
#pragma unroll
        for (int kk = 0; kk < 4; ++kk) {
            const int k0 = kk * 32;
            const v16h a  = load_a_frag(hbuf + m * Hd, k0, hi);
            const v16h bA = load_b_frag(whh, n0 + lm,      k0, hi);
            const v16h bB = load_b_frag(whh, n0 + 16 + lm, k0, hi);
            accA = __builtin_amdgcn_wmma_f32_16x16x32_f16(false, a, false, bA, (short)0, accA, false, false);
            accB = __builtin_amdgcn_wmma_f32_16x16x32_f16(false, a, false, bB, (short)0, accB, false, false);
        }

        // D layout: element r of lane -> gates[M = r + 8*hi][N = n0(+16) + lm]
#pragma unroll
        for (int r = 0; r < 8; ++r) {
            const int row = r + hi * 8;
            gates[row * G4 + n0 + lm]      = accA[r];
            gates[row * G4 + n0 + 16 + lm] = accB[r];
        }
        __syncthreads();

        // Elementwise gate math: 16*128 cells, 4 per thread (gate order i,f,g,o).
        {
            const int e0 = tid * 4;
            const int me = e0 >> 7;        // row in tile
            const int j0 = e0 & 127;       // hidden index
            const float* grow = gates + me * G4;
            const v4f gi = *(const v4f*)(grow + j0);
            const v4f gf = *(const v4f*)(grow + 128 + j0);
            const v4f gg = *(const v4f*)(grow + 256 + j0);
            const v4f go = *(const v4f*)(grow + 384 + j0);
            v4f c = *(const v4f*)(cbuf + me * Hd + j0);
#pragma unroll
            for (int q = 0; q < 4; ++q) {
                const float cv = fsig(gf[q]) * c[q] + fsig(gi[q]) * tanhf(gg[q]);
                c[q] = cv;
                const float hv = fsig(go[q]) * tanhf(cv);
                hbuf[me * Hd + j0 + q] = (half_t)hv;
                if (FINAL) {
                    // out (S,H,B): out[t][j][b]
                    yf[(size_t)t * Hd * Bx + (size_t)(j0 + q) * Bx + (b0 + me)] = hv;
                } else {
                    // inter-layer (S,B,H) f16
                    yh[(size_t)t * Bx * Hd + (size_t)(b0 + me) * Hd + (j0 + q)] = (half_t)hv;
                }
            }
            *(v4f*)(cbuf + me * Hd + j0) = c;
        }
        __syncthreads();
    }
}

extern "C" void kernel_launch(void* const* d_in, const int* in_sizes, int n_in,
                              void* d_out, int out_size, void* d_ws, size_t ws_size,
                              hipStream_t stream) {
    (void)in_sizes; (void)n_in; (void)out_size; (void)ws_size;

    const float* x    = (const float*)d_in[0];
    const float* Wih0 = (const float*)d_in[1];
    const float* Whh0 = (const float*)d_in[2];
    const float* bih0 = (const float*)d_in[3];
    const float* bhh0 = (const float*)d_in[4];
    const float* Wih1 = (const float*)d_in[5];
    const float* Whh1 = (const float*)d_in[6];
    const float* bih1 = (const float*)d_in[7];
    const float* bhh1 = (const float*)d_in[8];
    const float* Wih2 = (const float*)d_in[9];
    const float* Whh2 = (const float*)d_in[10];
    const float* bih2 = (const float*)d_in[11];
    const float* bhh2 = (const float*)d_in[12];

    half_t* buf0 = (half_t*)d_ws;                              // (S,B,H) f16
    half_t* buf1 = buf0 + (size_t)Sx * Bx * Hd;                // (S,B,H) f16
    float*  out  = (float*)d_out;                              // (S,H,B) f32

    static int attr_done = 0; // attribute set is idempotent & deterministic
    (void)attr_done;
    hipFuncSetAttribute((const void*)lstm_layer_kernel<float,  false>,
                        hipFuncAttributeMaxDynamicSharedMemorySize, SMEM_BYTES);
    hipFuncSetAttribute((const void*)lstm_layer_kernel<half_t, false>,
                        hipFuncAttributeMaxDynamicSharedMemorySize, SMEM_BYTES);
    hipFuncSetAttribute((const void*)lstm_layer_kernel<half_t, true >,
                        hipFuncAttributeMaxDynamicSharedMemorySize, SMEM_BYTES);

    const dim3 grid(Bx / ROWS);   // 32 workgroups, one 16-row batch tile each
    const dim3 block(TPB);        // 16 waves

    // Layer 0: x is (B,S,IN) f32 -> strides (S*IN, IN)
    lstm_layer_kernel<float, false><<<grid, block, SMEM_BYTES, stream>>>(
        x, (long long)Sx * Hd, (long long)Hd,
        Wih0, Whh0, bih0, bhh0, buf0, nullptr);

    // Layer 1: input (S,B,H) f16 -> strides (H, B*H)
    lstm_layer_kernel<half_t, false><<<grid, block, SMEM_BYTES, stream>>>(
        buf0, (long long)Hd, (long long)Bx * Hd,
        Wih1, Whh1, bih1, bhh1, buf1, nullptr);

    // Layer 2: final, writes (S,H,B) f32 directly
    lstm_layer_kernel<half_t, true><<<grid, block, SMEM_BYTES, stream>>>(
        buf1, (long long)Hd, (long long)Bx * Hd,
        Wih2, Whh2, bih2, bhh2, nullptr, out);
}
